// LSTMModel_5_7636451852952
// MI455X (gfx1250) — matrix operand order repaired and checked by reading, not compiled, not run on hardware
//
#include <hip/hip_runtime.h>
#include <cstdint>
#include <cstddef>

// Problem dims
#define B_    64
#define T_    512
#define IN_   256
#define HID_  512
#define G4_   2048     // 4*HID
#define OUT_  128
#define NWG_  16       // workgroups in persistent recurrent kernel

typedef __attribute__((ext_vector_type(16))) __bf16 v16bf;
typedef __attribute__((ext_vector_type(8)))  float  v8f;

// ---------- small helpers ----------
__device__ __forceinline__ unsigned short f2bf(float f) {
  union { float f; unsigned u; } v; v.f = f;
  unsigned r = v.u + 0x7FFFu + ((v.u >> 16) & 1u);   // round-nearest-even
  return (unsigned short)(r >> 16);
}
__device__ __forceinline__ float bf2f(unsigned short u) {
  union { unsigned u; float f; } v; v.u = ((unsigned)u) << 16;
  return v.f;
}
__device__ __forceinline__ float sigm(float x) { return 1.0f / (1.0f + __expf(-x)); }
__device__ __forceinline__ float tanh_fast(float x) {
  float e = __expf(2.0f * x);
  return (e - 1.0f) / (e + 1.0f);
}

union FragU { v16bf v; uint4 q[2]; };

// A-matrix 16x32 bf16 fragment (row-major source, row stride ldk elements).
// Lanes 0-15: M=lane, K base 0; lanes 16-31: M=lane-16, K base 8.
// Elements 0-7 -> K = kb+0..7 (16B contiguous), 8-15 -> K = kb+16..23 (16B contiguous).
__device__ __forceinline__ v16bf load_frag_a(const unsigned short* base, int ldk, int lane) {
  const unsigned short* p = base + (lane & 15) * ldk + ((lane >> 4) << 3);
  FragU u;
  u.q[0] = *(const uint4*)(p);
  u.q[1] = *(const uint4*)(p + 16);
  return u.v;
}
// B-matrix 32x16 bf16 fragment where B = W^T and W is row-major (N x K).
// Lane holds column n = lane&15 (one W row), kb = (lane<16 ? 0 : 8); elements 0-7 -> K = kb+0..7, 8-15 -> K = kb+16..23.
__device__ __forceinline__ v16bf load_frag_bT(const unsigned short* wrow, int ldk, int lane) {
  const unsigned short* p = wrow + (lane & 15) * ldk + ((lane >> 4) << 3);
  FragU u;
  u.q[0] = *(const uint4*)(p);
  u.q[1] = *(const uint4*)(p + 16);
  return u.v;
}

// ---- CDNA5 async global->LDS copy (ASYNCcnt path) ----
// Generic shared pointer: low 32 bits are the LDS byte offset (flat aperture rule),
// which is exactly what the instruction's VDST operand wants.
__device__ __forceinline__ void async_b128_to_lds(void* lds_ptr, const void* gptr) {
  unsigned lds_off = (unsigned)(uintptr_t)lds_ptr;
  asm volatile("global_load_async_to_lds_b128 %0, %1, off"
               :: "v"(lds_off), "v"(gptr)
               : "memory");
}
__device__ __forceinline__ void wait_asynccnt0() {
  asm volatile("s_wait_asynccnt 0x0" ::: "memory");
}

// ---------- conversion / init kernels ----------
__global__ void cvt_f32_bf16_kernel(const float* __restrict__ src,
                                    unsigned short* __restrict__ dst, int n) {
  int stride = gridDim.x * blockDim.x;
  for (int i = blockIdx.x * blockDim.x + threadIdx.x; i < n; i += stride)
    dst[i] = f2bf(src[i]);
}

__global__ void zero_u16_kernel(unsigned short* __restrict__ p, int n) {
  int stride = gridDim.x * blockDim.x;
  for (int i = blockIdx.x * blockDim.x + threadIdx.x; i < n; i += stride)
    p[i] = 0;
}

__global__ void bias_init_kernel(const float* bih0, const float* bhh0,
                                 const float* bih1, const float* bhh1,
                                 float* bias0, float* bias1, unsigned* cnt) {
  int i = blockIdx.x * blockDim.x + threadIdx.x;
  if (i < G4_) {
    bias0[i] = bih0[i] + bhh0[i];
    bias1[i] = bih1[i] + bhh1[i];
  }
  if (i < 4) cnt[i] = 0u;
}

// ---------- big bf16 WMMA GEMM: Out[M,N] = A[M,K] @ W[N,K]^T (bf16 out) ----------
// 64x128 tile per 256-thread workgroup (8 waves, 4 accumulators each).
// Software-pipelined: next k-step fragments load while current WMMAs issue.
__global__ void gemm_bf16_kernel(const unsigned short* __restrict__ A,
                                 const unsigned short* __restrict__ W,
                                 unsigned short* __restrict__ Out,
                                 int M, int N, int K) {
  const int nblocks = N >> 7;                 // N/128
  const int mblk = blockIdx.x / nblocks;
  const int nblk = blockIdx.x % nblocks;
  const int tid  = threadIdx.x;
  const int wave = tid >> 5;
  const int lane = tid & 31;

  const int m0 = mblk * 64 + (wave & 3) * 16;
  const int n0 = nblk * 128 + (wave >> 2) * 64;

  v8f acc[4] = {};
  const unsigned short* arow = A + (size_t)m0 * K;

  v16bf a_c = load_frag_a(arow, K, lane);
  v16bf b_c[4];
#pragma unroll
  for (int j = 0; j < 4; ++j)
    b_c[j] = load_frag_bT(W + (size_t)(n0 + 16 * j) * K, K, lane);

  for (int k = 0; k + 32 < K; k += 32) {
    // prefetch + preload next k-step while current WMMAs execute
    __builtin_prefetch((const void*)(arow + k + 64), 0, 1);
    v16bf a_n = load_frag_a(arow + k + 32, K, lane);
    v16bf b_n[4];
#pragma unroll
    for (int j = 0; j < 4; ++j)
      b_n[j] = load_frag_bT(W + (size_t)(n0 + 16 * j) * K + k + 32, K, lane);
#pragma unroll
    for (int j = 0; j < 4; ++j)
      acc[j] = __builtin_amdgcn_wmma_f32_16x16x32_bf16(
          false, a_c, false, b_c[j], (short)0, acc[j], false, false);
    a_c = a_n;
#pragma unroll
    for (int j = 0; j < 4; ++j) b_c[j] = b_n[j];
  }
#pragma unroll
  for (int j = 0; j < 4; ++j)
    acc[j] = __builtin_amdgcn_wmma_f32_16x16x32_bf16(
        false, a_c, false, b_c[j], (short)0, acc[j], false, false);

  const int row_off = (lane >> 4) << 3;       // 0 or 8
  const int col     = lane & 15;
#pragma unroll
  for (int j = 0; j < 4; ++j) {
#pragma unroll
    for (int r = 0; r < 8; ++r) {
      int row = m0 + r + row_off;
      Out[(size_t)row * N + n0 + 16 * j + col] = f2bf(acc[j][r]);
    }
  }
}

// ---------- persistent LSTM recurrence ----------
// 16 WGs; WG g owns hidden columns [32g, 32g+32) => gate columns {q*512 + 32g .. +32}, q=0..3.
// Per step: h(64x512) streamed through LDS in 4KB chunks with async-to-LDS (double
// buffered), gates = h @ Whh_slice^T via WMMA (A from LDS, B from L2-resident W_hh),
// +xg +bias, activations, cell state in registers, double-buffered global h,
// monotonic-counter grid barrier.
__global__ void lstm_rec_kernel(const unsigned short* __restrict__ xg,    // [B*T, 2048] rows bt=b*T+t
                                const unsigned short* __restrict__ Whh,   // [2048, 512] bf16
                                const float* __restrict__ bias,           // [2048] combined
                                unsigned short* __restrict__ hbuf,        // [2][64*512] bf16
                                unsigned short* __restrict__ hs,          // [B*T, 512] bf16 out
                                unsigned* __restrict__ cnt) {
  __shared__ float gates[64 * 128];                  // 32 KB pre-activations
  __shared__ unsigned short hstage[2][64 * 32];      // 2 x 4 KB async-staged h chunks

  const int tid  = threadIdx.x;
  const int wave = tid >> 5;
  const int lane = tid & 31;
  const int wg   = blockIdx.x;                 // 0..15

  const int m0 = (wave & 3) * 16;              // batch tile
  const int ng = (wave >> 2) << 2;             // first of this wave's 4 n-tiles
  const int row_off = (lane >> 4) << 3;
  const int col = lane & 15;

  int grow[4];                                 // W_hh row base per n-tile (wave-constant)
#pragma unroll
  for (int j = 0; j < 4; ++j) {
    int nt = ng + j;
    grow[j] = (nt >> 1) * HID_ + wg * 32 + (nt & 1) * 16;
  }
  // per-thread async chunk mapping: one b128 per thread per chunk
  const int crow = tid >> 2;                   // h row 0..63
  const int csub = (tid & 3) * 8;              // 8 bf16 within the 32-wide chunk

  float cstate[8];
#pragma unroll
  for (int e = 0; e < 8; ++e) cstate[e] = 0.0f;

  for (int t = 0; t < T_; ++t) {
    const unsigned short* hread  = hbuf + (size_t)(t & 1) * (B_ * HID_);
    unsigned short*       hwrite = hbuf + (size_t)((t + 1) & 1) * (B_ * HID_);

    // prologue: stream chunk 0 into buffer 0
    async_b128_to_lds(&hstage[0][tid * 8], hread + (size_t)crow * HID_ + csub);

    v8f acc[4] = {};
    for (int ki = 0; ki < HID_ / 32; ++ki) {
      const int k  = ki * 32;
      const int buf = ki & 1;
      wait_asynccnt0();                        // this wave's chunk ki has landed
      __syncthreads();                         // everyone's chunk ki has landed
      if (k + 32 < HID_)                       // stream chunk ki+1 into other buffer
        async_b128_to_lds(&hstage[buf ^ 1][tid * 8],
                          hread + (size_t)crow * HID_ + (k + 32) + csub);

      const unsigned short* hsp = (const unsigned short*)hstage[buf];
      v16bf a  = load_frag_a(hsp + m0 * 32, 32, lane);        // ds_load_b128 path
      v16bf b0 = load_frag_bT(Whh + (size_t)grow[0] * HID_ + k, HID_, lane);
      v16bf b1 = load_frag_bT(Whh + (size_t)grow[1] * HID_ + k, HID_, lane);
      v16bf b2 = load_frag_bT(Whh + (size_t)grow[2] * HID_ + k, HID_, lane);
      v16bf b3 = load_frag_bT(Whh + (size_t)grow[3] * HID_ + k, HID_, lane);
      acc[0] = __builtin_amdgcn_wmma_f32_16x16x32_bf16(false, a, false, b0, (short)0, acc[0], false, false);
      acc[1] = __builtin_amdgcn_wmma_f32_16x16x32_bf16(false, a, false, b1, (short)0, acc[1], false, false);
      acc[2] = __builtin_amdgcn_wmma_f32_16x16x32_bf16(false, a, false, b2, (short)0, acc[2], false, false);
      acc[3] = __builtin_amdgcn_wmma_f32_16x16x32_bf16(false, a, false, b3, (short)0, acc[3], false, false);
    }

    // stage pre-activations into LDS, local layout [batch][gate*32 + c]
    __syncthreads();
#pragma unroll
    for (int j = 0; j < 4; ++j) {
      int nt = ng + j;
      int ln = (nt >> 1) * 32 + (nt & 1) * 16 + col;
#pragma unroll
      for (int r = 0; r < 8; ++r)
        gates[(m0 + r + row_off) * 128 + ln] = acc[j][r];
    }
    __syncthreads();

    // --- fused gate activations + cell/h update ---
#pragma unroll
    for (int e = 0; e < 8; ++e) {
      int flat = tid + e * 256;                // 0..2047 == 64 batches x 32 cols
      int b_ = flat >> 5;
      int c_ = flat & 31;
      int gcol = wg * 32 + c_;                 // hidden column (global)
      size_t xrow = (size_t)(b_ * T_ + t) * G4_;
      float ri = gates[b_ * 128 +  0 + c_] + bf2f(xg[xrow + 0 * HID_ + gcol]) + bias[0 * HID_ + gcol];
      float rf = gates[b_ * 128 + 32 + c_] + bf2f(xg[xrow + 1 * HID_ + gcol]) + bias[1 * HID_ + gcol];
      float rg = gates[b_ * 128 + 64 + c_] + bf2f(xg[xrow + 2 * HID_ + gcol]) + bias[2 * HID_ + gcol];
      float ro = gates[b_ * 128 + 96 + c_] + bf2f(xg[xrow + 3 * HID_ + gcol]) + bias[3 * HID_ + gcol];
      float ig = sigm(ri), fg = sigm(rf), gg = tanh_fast(rg), og = sigm(ro);
      cstate[e] = fg * cstate[e] + ig * gg;
      float h = og * tanh_fast(cstate[e]);
      unsigned short hb = f2bf(h);
      hwrite[(size_t)b_ * HID_ + gcol] = hb;
      hs[(size_t)(b_ * T_ + t) * HID_ + gcol] = hb;
    }

    // --- grid-wide barrier (monotonic counter) ---
    __syncthreads();
    if (tid == 0) {
      __threadfence();
      atomicAdd(cnt, 1u);
      unsigned target = (unsigned)(t + 1) * NWG_;
      while (atomicAdd(cnt, 0u) < target) { __builtin_amdgcn_s_sleep(2); }
      __threadfence();
    }
    __syncthreads();
  }
}

// ---------- final FC: out[b,o] = hs1[b,T-1,:] . fc_w[o,:] + fc_b[o] ----------
__global__ void fc_kernel(const unsigned short* __restrict__ hs1,
                          const float* __restrict__ fcw,
                          const float* __restrict__ fcb,
                          float* __restrict__ out) {
  int idx = blockIdx.x * blockDim.x + threadIdx.x;
  if (idx >= B_ * OUT_) return;
  int b = idx >> 7;
  int o = idx & (OUT_ - 1);
  const unsigned short* hp = hs1 + (size_t)(b * T_ + (T_ - 1)) * HID_;
  const float* wp = fcw + (size_t)o * HID_;
  float s = fcb[o];
  for (int k = 0; k < HID_; ++k) s += bf2f(hp[k]) * wp[k];
  out[idx] = s;
}

// ---------- workspace layout (bytes) ----------
static const size_t OFF_XG   = 0;                                   // 128 MB (reused L0/L1)
static const size_t OFF_HS0  = OFF_XG   + (size_t)B_*T_*G4_*2;      // 32 MB
static const size_t OFF_HS1  = OFF_HS0  + (size_t)B_*T_*HID_*2;     // 32 MB
static const size_t OFF_XBF  = OFF_HS1  + (size_t)B_*T_*HID_*2;     // 16 MB
static const size_t OFF_WIH0 = OFF_XBF  + (size_t)B_*T_*IN_*2;      // 1 MB
static const size_t OFF_WHH0 = OFF_WIH0 + (size_t)G4_*IN_*2;        // 2 MB
static const size_t OFF_WIH1 = OFF_WHH0 + (size_t)G4_*HID_*2;       // 2 MB
static const size_t OFF_WHH1 = OFF_WIH1 + (size_t)G4_*HID_*2;       // 2 MB
static const size_t OFF_HBUF = OFF_WHH1 + (size_t)G4_*HID_*2;       // 128 KB
static const size_t OFF_B0   = OFF_HBUF + (size_t)2*B_*HID_*2;      // 8 KB
static const size_t OFF_B1   = OFF_B0   + (size_t)G4_*4;            // 8 KB
static const size_t OFF_CNT  = OFF_B1   + (size_t)G4_*4;            // counters

extern "C" void kernel_launch(void* const* d_in, const int* in_sizes, int n_in,
                              void* d_out, int out_size, void* d_ws, size_t ws_size,
                              hipStream_t stream) {
  (void)in_sizes; (void)n_in; (void)out_size; (void)ws_size;

  const float* x    = (const float*)d_in[0];
  const float* Wih0 = (const float*)d_in[1];
  const float* Whh0 = (const float*)d_in[2];
  const float* bih0 = (const float*)d_in[3];
  const float* bhh0 = (const float*)d_in[4];
  const float* Wih1 = (const float*)d_in[5];
  const float* Whh1 = (const float*)d_in[6];
  const float* bih1 = (const float*)d_in[7];
  const float* bhh1 = (const float*)d_in[8];
  const float* fcw  = (const float*)d_in[9];
  const float* fcb  = (const float*)d_in[10];
  float* out = (float*)d_out;

  char* ws = (char*)d_ws;
  unsigned short* xg    = (unsigned short*)(ws + OFF_XG);
  unsigned short* hs0   = (unsigned short*)(ws + OFF_HS0);
  unsigned short* hs1   = (unsigned short*)(ws + OFF_HS1);
  unsigned short* xbf   = (unsigned short*)(ws + OFF_XBF);
  unsigned short* wih0b = (unsigned short*)(ws + OFF_WIH0);
  unsigned short* whh0b = (unsigned short*)(ws + OFF_WHH0);
  unsigned short* wih1b = (unsigned short*)(ws + OFF_WIH1);
  unsigned short* whh1b = (unsigned short*)(ws + OFF_WHH1);
  unsigned short* hbuf  = (unsigned short*)(ws + OFF_HBUF);
  float*          bias0 = (float*)(ws + OFF_B0);
  float*          bias1 = (float*)(ws + OFF_B1);
  unsigned*       cnt   = (unsigned*)(ws + OFF_CNT);

  dim3 blk(256);

  // 1) bf16 conversions
  cvt_f32_bf16_kernel<<<2048, blk, 0, stream>>>(x,    xbf,   B_*T_*IN_);
  cvt_f32_bf16_kernel<<<512,  blk, 0, stream>>>(Wih0, wih0b, G4_*IN_);
  cvt_f32_bf16_kernel<<<512,  blk, 0, stream>>>(Whh0, whh0b, G4_*HID_);
  cvt_f32_bf16_kernel<<<512,  blk, 0, stream>>>(Wih1, wih1b, G4_*HID_);
  cvt_f32_bf16_kernel<<<512,  blk, 0, stream>>>(Whh1, whh1b, G4_*HID_);
  bias_init_kernel<<<8, blk, 0, stream>>>(bih0, bhh0, bih1, bhh1, bias0, bias1, cnt);
  zero_u16_kernel<<<256, blk, 0, stream>>>(hbuf, 2 * B_ * HID_);

  // 2) layer 0: xg = x @ Wih0^T  (M=32768, N=2048, K=256)
  gemm_bf16_kernel<<<(B_*T_/64) * (G4_/128), blk, 0, stream>>>(
      xbf, wih0b, xg, B_*T_, G4_, IN_);
  // 3) layer 0 recurrence
  lstm_rec_kernel<<<NWG_, blk, 0, stream>>>(xg, whh0b, bias0, hbuf, hs0, cnt);

  // 4) layer 1: re-zero h, xg = hs0 @ Wih1^T (K=512), recurrence with cnt[1]
  zero_u16_kernel<<<256, blk, 0, stream>>>(hbuf, 2 * B_ * HID_);
  gemm_bf16_kernel<<<(B_*T_/64) * (G4_/128), blk, 0, stream>>>(
      hs0, wih1b, xg, B_*T_, G4_, HID_);
  lstm_rec_kernel<<<NWG_, blk, 0, stream>>>(xg, whh1b, bias1, hbuf, hs1, cnt + 1);

  // 5) FC head
  fc_kernel<<<(B_*OUT_ + 255) / 256, blk, 0, stream>>>(hs1, fcw, fcb, out);
}
